// TPHunyuanMoE_65060164600411
// MI455X (gfx1250) — matrix operand order repaired and checked by reading, not compiled, not run on hardware
//
#include <hip/hip_runtime.h>
#include <hip/hip_bf16.h>

typedef __attribute__((ext_vector_type(16))) __bf16 v16bf;
typedef __attribute__((ext_vector_type(8)))  __bf16 v8bf;
typedef __attribute__((ext_vector_type(8)))  float  v8f;

#define HIDDEN       4096
#define MOE_INTER    1024
#define SHARED_INTER 4096
#define NUM_EXPERTS  16
#define TOKENS       4096   // 2 * 2048
#define TOPK         2

// ---------------------------------------------------------------- cast fp32 -> bf16
__global__ void cast_f32_bf16_kernel(const float* __restrict__ in,
                                     __bf16* __restrict__ out, int n4) {
  int i = blockIdx.x * blockDim.x + threadIdx.x;
  if (i >= n4) return;
  float4 v = ((const float4*)in)[i];
  union { __bf16 b[4]; uint2 u; } t;
  t.b[0] = (__bf16)v.x; t.b[1] = (__bf16)v.y; t.b[2] = (__bf16)v.z; t.b[3] = (__bf16)v.w;
  *(uint2*)(out + (size_t)i * 4) = t.u;
}

// ---------------------------------------------------------------- router: softmax + top2
__global__ void router_kernel(const float* __restrict__ x, const float* __restrict__ gw,
                              int* __restrict__ tk_idx, float* __restrict__ tk_w) {
  int lane = threadIdx.x & 31;
  int wave = threadIdx.x >> 5;
  int t = blockIdx.x * (blockDim.x >> 5) + wave;
  if (t >= TOKENS) return;
  float acc[NUM_EXPERTS];
#pragma unroll
  for (int e = 0; e < NUM_EXPERTS; e++) acc[e] = 0.f;
  const float* xr = x + (size_t)t * HIDDEN;
  for (int h = lane; h < HIDDEN; h += 32) {
    float xv = xr[h];
#pragma unroll
    for (int e = 0; e < NUM_EXPERTS; e++) acc[e] = fmaf(xv, gw[e * HIDDEN + h], acc[e]);
  }
#pragma unroll
  for (int e = 0; e < NUM_EXPERTS; e++)
    for (int off = 16; off > 0; off >>= 1) acc[e] += __shfl_xor(acc[e], off, 32);
  float mx = acc[0];
#pragma unroll
  for (int e = 1; e < NUM_EXPERTS; e++) mx = fmaxf(mx, acc[e]);
  float p[NUM_EXPERTS];
#pragma unroll
  for (int e = 0; e < NUM_EXPERTS; e++) p[e] = __expf(acc[e] - mx);
  int i0 = 0; float v0 = p[0];
#pragma unroll
  for (int e = 1; e < NUM_EXPERTS; e++) if (p[e] > v0) { v0 = p[e]; i0 = e; }
  int i1 = (i0 == 0) ? 1 : 0; float v1 = p[i1];
#pragma unroll
  for (int e = 0; e < NUM_EXPERTS; e++)
    if (e != i0 && p[e] > v1) { v1 = p[e]; i1 = e; }
  if (lane == 0) {
    float inv = 1.f / (v0 + v1);
    tk_idx[2 * t] = i0;  tk_idx[2 * t + 1] = i1;
    tk_w[2 * t] = v0 * inv;
    tk_w[2 * t + 1] = v1 * inv;
  }
}

// ---------------------------------------------------------------- deterministic expert lists
__global__ void build_lists_kernel(const int* __restrict__ tk_idx,
                                   int* __restrict__ counts, int* __restrict__ offsets,
                                   int* __restrict__ tok_list, int* __restrict__ slot_of) {
  __shared__ int sc[NUM_EXPERTS];
  __shared__ int so[NUM_EXPERTS + 1];
  const int lane = threadIdx.x & 31;
  const int e = threadIdx.x >> 5;           // 16 waves, one expert each
  const int TK = TOKENS * TOPK;             // multiple of 32 -> uniform ballot loops
  int cnt = 0;
  for (int i = lane; i < TK; i += 32) {
    unsigned m = __builtin_amdgcn_ballot_w32(tk_idx[i] == e);
    cnt += __popc(m);
  }
  if (lane == 0) sc[e] = cnt;
  __syncthreads();
  if (threadIdx.x == 0) {
    int s = 0;
    for (int k = 0; k < NUM_EXPERTS; k++) { so[k] = s; s += sc[k]; }
    so[NUM_EXPERTS] = s;
  }
  __syncthreads();
  int base = so[e];
  for (int i = lane; i < TK; i += 32) {
    bool match = (tk_idx[i] == e);
    unsigned m = __builtin_amdgcn_ballot_w32(match);
    int rank = __popc(m & ((1u << lane) - 1u));
    if (match) {
      int slot = base + rank;
      tok_list[slot] = i >> 1;   // token id
      slot_of[i] = slot;
    }
    base += __popc(m);
  }
  if (lane == 0) {
    counts[e] = sc[e];
    offsets[e] = so[e];
    if (e == 0) offsets[NUM_EXPERTS] = so[NUM_EXPERTS];
  }
}

// ---------------------------------------------------------------- tiled bf16 WMMA GEMM
// C[M,N] = A[M,K] (bf16, optionally row-gathered) * W[N,K]^T (fp32 rows -> bf16 in LDS).
// Double-buffered LDS; A tiles staged with GLOBAL_LOAD_ASYNC_TO_LDS_B128 (ASYNCcnt),
// W tiles staged through VGPRs for fp32->bf16 conversion, overlapped with WMMA.
// SWIGLU: W has 2N rows (gate=n, up=n+N); epilogue writes silu(g)*u.
// counts != null => per-expert mode (blockIdx.z = expert id).
template <bool SWIGLU, bool GATHER, bool OUTF32>
__global__ __launch_bounds__(256)
void gemm_bf16_wmma(const __bf16* __restrict__ A, const float* __restrict__ Wbase,
                    void* __restrict__ OutV,
                    const int* __restrict__ counts, const int* __restrict__ offsets,
                    const int* __restrict__ tok_list,
                    int M, int N, int K, long wstride) {
  constexpr int BM = 128, BN = 64, BK = 32, LDT = BK + 8;
  __shared__ __bf16 As[2][BM][LDT];
  __shared__ __bf16 Bg[2][BN][LDT];
  __shared__ __bf16 Bu[2][SWIGLU ? BN : 1][LDT];

  const int e = blockIdx.z;
  int Me = M, rowbase = 0;
  const float* W = Wbase;
  if (counts) { Me = counts[e]; rowbase = offsets[e]; W = Wbase + (long)e * wstride; }
  const int m0 = blockIdx.y * BM;
  if (m0 >= Me) return;
  const int n0 = blockIdx.x * BN;

  const int tid = threadIdx.x;
  const int lane = tid & 31;
  const int wave = tid >> 5;          // 8 waves
  const int wm = wave >> 1;           // 0..3 (M)
  const int wn = wave & 1;            // 0..1 (N)
  const int laneHi = lane >> 4;
  const int lane15 = lane & 15;

  union Frag { v16bf v; v8bf h[2]; };
  const v8f vzero = {0.f, 0.f, 0.f, 0.f, 0.f, 0.f, 0.f, 0.f};
  v8f cg[2][2], cu[2][2];
#pragma unroll
  for (int i = 0; i < 2; i++)
#pragma unroll
    for (int j = 0; j < 2; j++) { cg[i][j] = vzero; cu[i][j] = vzero; }

  // ---- per-thread loader geometry
  const int lrow = tid >> 2;  // 0..63
  const int seg = tid & 3;    // 0..3 (8-element, 16B k-chunks)

  // A: two rows per thread; precompute global element offsets (k0 added later)
  long arow_g[2];
#pragma unroll
  for (int r2 = 0; r2 < 2; r2++) {
    int lr = m0 + lrow + r2 * 64;
    int lrc = lr < Me ? lr : (Me - 1);
    long grow;
    if (GATHER)      grow = tok_list[rowbase + lrc];
    else if (counts) grow = (long)rowbase + lrc;
    else             grow = lr < M ? lr : (long)(M - 1);
    arow_g[r2] = grow * (long)K + seg * 8;
  }
  const float* wg_ptr = W + (long)(n0 + lrow) * K + seg * 8;
  const float* wu_ptr = SWIGLU ? (W + (long)(n0 + lrow + N) * K + seg * 8) : wg_ptr;

  // async A-tile stage: global -> LDS, no VGPR round-trip (tracked by ASYNCcnt)
  auto async_a = [&](int k0, int buf) {
#pragma unroll
    for (int r2 = 0; r2 < 2; r2++) {
      unsigned lds = (unsigned)(uintptr_t)&As[buf][lrow + r2 * 64][seg * 8];
      unsigned long long ga = (unsigned long long)(uintptr_t)(A + arow_g[r2] + k0);
      asm volatile("global_load_async_to_lds_b128 %0, %1, off"
                   :: "v"(lds), "v"(ga) : "memory");
    }
  };

  float4 wf0, wf1, uf0, uf1;
  auto load_w = [&](int k0) {
    wf0 = *(const float4*)(wg_ptr + k0);
    wf1 = *(const float4*)(wg_ptr + k0 + 4);
    if (SWIGLU) {
      uf0 = *(const float4*)(wu_ptr + k0);
      uf1 = *(const float4*)(wu_ptr + k0 + 4);
    }
  };
  auto store_w = [&](int buf) {
    union { __bf16 b[8]; uint4 u; } t;
    t.b[0] = (__bf16)wf0.x; t.b[1] = (__bf16)wf0.y; t.b[2] = (__bf16)wf0.z; t.b[3] = (__bf16)wf0.w;
    t.b[4] = (__bf16)wf1.x; t.b[5] = (__bf16)wf1.y; t.b[6] = (__bf16)wf1.z; t.b[7] = (__bf16)wf1.w;
    *(uint4*)&Bg[buf][lrow][seg * 8] = t.u;
    if (SWIGLU) {
      union { __bf16 b[8]; uint4 u; } s;
      s.b[0] = (__bf16)uf0.x; s.b[1] = (__bf16)uf0.y; s.b[2] = (__bf16)uf0.z; s.b[3] = (__bf16)uf0.w;
      s.b[4] = (__bf16)uf1.x; s.b[5] = (__bf16)uf1.y; s.b[6] = (__bf16)uf1.z; s.b[7] = (__bf16)uf1.w;
      *(uint4*)&Bu[buf][lrow][seg * 8] = s.u;
    }
  };

  // ---- prologue: stage tile 0
  async_a(0, 0);
  load_w(0);
  store_w(0);
  asm volatile("s_wait_asynccnt 0x0" ::: "memory");

  const int nk = K / BK;
  int cur = 0;
  for (int it = 0; it < nk; ++it) {
    __syncthreads();  // tile `it` visible in LDS[cur] for the whole block
    const int nxt = cur ^ 1;
    const bool more = (it + 1 < nk);
    if (more) {
      async_a((it + 1) * BK, nxt);       // A: async global->LDS, overlaps WMMA
      load_w((it + 1) * BK);             // W: fp32 loads in flight during WMMA
      if (it + 2 < nk) {
        __builtin_prefetch(wg_ptr + (it + 2) * BK, 0, 1);   // global_prefetch_b8
        if (SWIGLU) __builtin_prefetch(wu_ptr + (it + 2) * BK, 0, 1);
      }
    }

    // ---- fragments per ISA wave32 layouts + WMMA
    Frag a[2], bg[2], bu[2];
#pragma unroll
    for (int mt = 0; mt < 2; mt++) {
      int ar = wm * 32 + mt * 16 + lane15;     // A: M = lane&15
      int kb = laneHi * 8;
      a[mt].h[0] = *(const v8bf*)&As[cur][ar][kb];
      a[mt].h[1] = *(const v8bf*)&As[cur][ar][kb + 16];
    }
#pragma unroll
    for (int nt = 0; nt < 2; nt++) {
      int bc = wn * 32 + nt * 16 + lane15;     // B: N = lane&15
      int kh = laneHi * 8;
      bg[nt].h[0] = *(const v8bf*)&Bg[cur][bc][kh];
      bg[nt].h[1] = *(const v8bf*)&Bg[cur][bc][kh + 16];
      if (SWIGLU) {
        bu[nt].h[0] = *(const v8bf*)&Bu[cur][bc][kh];
        bu[nt].h[1] = *(const v8bf*)&Bu[cur][bc][kh + 16];
      }
    }
#pragma unroll
    for (int mt = 0; mt < 2; mt++)
#pragma unroll
      for (int nt = 0; nt < 2; nt++) {
        cg[mt][nt] = __builtin_amdgcn_wmma_f32_16x16x32_bf16(
            false, a[mt].v, false, bg[nt].v, (short)0, cg[mt][nt], false, false);
        if (SWIGLU)
          cu[mt][nt] = __builtin_amdgcn_wmma_f32_16x16x32_bf16(
              false, a[mt].v, false, bu[nt].v, (short)0, cu[mt][nt], false, false);
      }

    if (more) {
      store_w(nxt);                                     // cvt + ds_store after WMMA issue
      asm volatile("s_wait_asynccnt 0x0" ::: "memory"); // this wave's A-tile landed
    }
    cur = nxt;
  }

  // ---- epilogue (C layout: VGPR i -> M = i + 8*(lane>=16), N = lane&15)
#pragma unroll
  for (int mt = 0; mt < 2; mt++)
#pragma unroll
    for (int nt = 0; nt < 2; nt++) {
      int col = n0 + wn * 32 + nt * 16 + lane15;
#pragma unroll
      for (int i = 0; i < 8; i++) {
        int lr = m0 + wm * 32 + mt * 16 + i + laneHi * 8;
        if (lr >= Me) continue;
        float val;
        if (SWIGLU) {
          float g = cg[mt][nt][i];
          float u = cu[mt][nt][i];
          val = (g / (1.f + __expf(-g))) * u;   // silu(g)*u
        } else {
          val = cg[mt][nt][i];
        }
        long orow = (long)rowbase + lr;
        if (OUTF32) ((float*)OutV)[orow * (long)N + col] = val;
        else        ((__bf16*)OutV)[orow * (long)N + col] = (__bf16)val;
      }
    }
}

// ---------------------------------------------------------------- combine: out += w0*e0 + w1*e1
__global__ void combine_kernel(float* __restrict__ out, const __bf16* __restrict__ dn,
                               const int* __restrict__ slot_of,
                               const float* __restrict__ tk_w, int n4) {
  int i = blockIdx.x * blockDim.x + threadIdx.x;
  if (i >= n4) return;
  const int per = HIDDEN / 4;
  int t = i / per;
  int h = (i % per) * 4;
  int s0 = slot_of[2 * t], s1 = slot_of[2 * t + 1];
  float w0 = tk_w[2 * t], w1 = tk_w[2 * t + 1];
  const __bf16* d0 = dn + (long)s0 * HIDDEN + h;
  const __bf16* d1 = dn + (long)s1 * HIDDEN + h;
  float4 o = ((float4*)out)[i];
  o.x += w0 * (float)d0[0] + w1 * (float)d1[0];
  o.y += w0 * (float)d0[1] + w1 * (float)d1[1];
  o.z += w0 * (float)d0[2] + w1 * (float)d1[2];
  o.w += w0 * (float)d0[3] + w1 * (float)d1[3];
  ((float4*)out)[i] = o;
}

// ---------------------------------------------------------------- launcher
extern "C" void kernel_launch(void* const* d_in, const int* in_sizes, int n_in,
                              void* d_out, int out_size, void* d_ws, size_t ws_size,
                              hipStream_t stream) {
  const float* x      = (const float*)d_in[0];  // [T, H]
  const float* gate_w = (const float*)d_in[1];  // [E, H]
  const float* w_gu   = (const float*)d_in[2];  // [E, 2I, H]
  const float* w_dn   = (const float*)d_in[3];  // [E, H, I]
  const float* sw_gu  = (const float*)d_in[4];  // [2Is, H]
  const float* sw_dn  = (const float*)d_in[5];  // [H, Is]
  float* out = (float*)d_out;

  char* p = (char*)d_ws;
  auto take = [&](size_t bytes) { char* r = p; p += (bytes + 255) & ~(size_t)255; return r; };
  __bf16* x_bf   = (__bf16*)take((size_t)TOKENS * HIDDEN * 2);            // 32 MB
  __bf16* act_s  = (__bf16*)take((size_t)TOKENS * SHARED_INTER * 2);      // 32 MB
  __bf16* act_e  = (__bf16*)take((size_t)TOKENS * TOPK * MOE_INTER * 2);  // 16 MB
  __bf16* dn_out = (__bf16*)take((size_t)TOKENS * TOPK * HIDDEN * 2);     // 64 MB
  int*   tk_idx  = (int*)take((size_t)TOKENS * TOPK * 4);
  float* tk_w    = (float*)take((size_t)TOKENS * TOPK * 4);
  int*   counts  = (int*)take(NUM_EXPERTS * 4);
  int*   offsets = (int*)take((NUM_EXPERTS + 1) * 4);
  int*   tok_lst = (int*)take((size_t)TOKENS * TOPK * 4);
  int*   slot_of = (int*)take((size_t)TOKENS * TOPK * 4);
  (void)ws_size; (void)in_sizes; (void)n_in; (void)out_size;

  const int n4 = TOKENS * HIDDEN / 4;
  cast_f32_bf16_kernel<<<n4 / 256, 256, 0, stream>>>(x, x_bf, n4);
  router_kernel<<<TOKENS / 8, 256, 0, stream>>>(x, gate_w, tk_idx, tk_w);
  build_lists_kernel<<<1, NUM_EXPERTS * 32, 0, stream>>>(tk_idx, counts, offsets, tok_lst, slot_of);

  // shared SwiGLU: act_s[T, Is] (bf16)
  gemm_bf16_wmma<true, false, false><<<dim3(SHARED_INTER / 64, TOKENS / 128, 1), 256, 0, stream>>>(
      x_bf, sw_gu, act_s, nullptr, nullptr, nullptr, TOKENS, SHARED_INTER, HIDDEN, 0);
  // shared down: out[T, H] (fp32 base of result)
  gemm_bf16_wmma<false, false, true><<<dim3(HIDDEN / 64, TOKENS / 128, 1), 256, 0, stream>>>(
      act_s, sw_dn, out, nullptr, nullptr, nullptr, TOKENS, HIDDEN, SHARED_INTER, 0);
  // expert SwiGLU (gathered rows): act_e[slots, I] (bf16)
  gemm_bf16_wmma<true, true, false><<<dim3(MOE_INTER / 64, TOKENS / 128, NUM_EXPERTS), 256, 0, stream>>>(
      x_bf, w_gu, act_e, counts, offsets, tok_lst, TOKENS, MOE_INTER, HIDDEN,
      (long)2 * MOE_INTER * HIDDEN);
  // expert down: dn_out[slots, H] (bf16)
  gemm_bf16_wmma<false, false, false><<<dim3(HIDDEN / 64, TOKENS / 128, NUM_EXPERTS), 256, 0, stream>>>(
      act_e, w_dn, dn_out, counts, offsets, nullptr, TOKENS, HIDDEN, MOE_INTER,
      (long)HIDDEN * MOE_INTER);
  // out += w0*dn[s0] + w1*dn[s1]
  combine_kernel<<<n4 / 256, 256, 0, stream>>>(out, dn_out, slot_of, tk_w, n4);
}
